// _ExpertPair_5763846111360
// MI455X (gfx1250) — compile-verified, matrix operand order checked by reading, not run on hardware
//
#include <hip/hip_runtime.h>
#include <hip/hip_fp16.h>

typedef __attribute__((ext_vector_type(16))) _Float16 v16h;
typedef __attribute__((ext_vector_type(8)))  _Float16 v8h;
typedef __attribute__((ext_vector_type(4)))  _Float16 v4h;
typedef __attribute__((ext_vector_type(8)))  float    v8f;
typedef __attribute__((ext_vector_type(4)))  int      av4i;

#define T_TOK 16384
#define D_IN  1024
#define D_MID 4096
#define D_OUT 1024

#define BM 128
#define BN 128
#define BK 64
#define LDSS 72   // padded row stride (f16) : 144B = 36 banks -> conflict-free b128 frags

#if defined(__has_builtin)
#if __has_builtin(__builtin_amdgcn_global_load_async_to_lds_b128) && \
    __has_builtin(__builtin_amdgcn_s_wait_asynccnt)
#define USE_ASYNC 1
#else
#define USE_ASYNC 0
#endif
#else
#define USE_ASYNC 0
#endif

// Builtin expects int4* in AS1 (global) / AS3 (LDS); C++ has no implicit void*->T*.
#define GPTR(p) ((__attribute__((address_space(1))) av4i*)(uintptr_t)(p))
#define LPTR(p) ((__attribute__((address_space(3))) av4i*)(unsigned)(uintptr_t)(p))

__device__ __forceinline__ void async_fence() {
#if USE_ASYNC
  __builtin_amdgcn_s_wait_asynccnt(0);
#endif
}

// ---------------- x : f32 -> f16 ----------------
__global__ void k_cvt_f16(const float* __restrict__ x, _Float16* __restrict__ xh) {
  int i = (blockIdx.x * blockDim.x + threadIdx.x) * 4;
  float4 v = *reinterpret_cast<const float4*>(x + i);
  v4h o;
  o.x = (_Float16)v.x; o.y = (_Float16)v.y;
  o.z = (_Float16)v.z; o.w = (_Float16)v.w;
  *reinterpret_cast<v4h*>(xh + i) = o;
}

// ------ weight prep: Weff = (wq * scale[o]) @ Hb * signs, f16 output ------
__global__ void k_prep_w(const float* __restrict__ wq, const float* __restrict__ scale,
                         const float* __restrict__ signs, _Float16* __restrict__ weff,
                         int I) {
  __shared__ float s[128];
  const int nblk = I >> 7;
  const int o   = blockIdx.x / nblk;
  const int blk = blockIdx.x % nblk;
  const int t   = threadIdx.x;
  const size_t base = (size_t)o * I + ((size_t)blk << 7);
  s[t] = wq[base + t];
  __syncthreads();
#pragma unroll
  for (int len = 1; len < 128; len <<= 1) {
    float a = s[t];
    float b = s[t ^ len];
    __syncthreads();
    s[t] = (t & len) ? (b - a) : (a + b);
    __syncthreads();
  }
  float y = s[t] * scale[o] * signs[(blk << 7) + t] * 0.08838834764831845f; // 1/sqrt(128)
  weff[base + t] = (_Float16)y;
}

// ---------------- fast tanh-GELU (branch-free) ----------------
__device__ __forceinline__ float fast_gelu(float v) {
  float u = 0.7978845608028654f * (v + 0.044715f * v * v * v);
#if defined(__has_builtin) && __has_builtin(__builtin_amdgcn_tanhf)
  float th = __builtin_amdgcn_tanhf(u);
#else
  // arg of exp is always <= 0: no overflow, no divergence
  float e  = __expf(-2.0f * __builtin_fabsf(u));
  float th = __builtin_copysignf((1.0f - e) * __builtin_amdgcn_rcpf(1.0f + e), u);
#endif
  return 0.5f * v * (1.0f + th);
}

// ---- stage a 128x64 f16 tile of A and of B into LDS (async when available) ----
__device__ __forceinline__ void copy_tiles(const _Float16* __restrict__ A,
                                           const _Float16* __restrict__ B,
                                           _Float16* sA, _Float16* sB,
                                           int m0, int n0, int kt, int K, int tid) {
  const uint4* gA = reinterpret_cast<const uint4*>(A + (size_t)m0 * K + kt);
  const uint4* gB = reinterpret_cast<const uint4*>(B + (size_t)n0 * K + kt);
  const int ldg = K >> 3;  // row stride in uint4
#pragma unroll
  for (int i = 0; i < 4; i++) {
    int c = tid + (i << 8);  // chunk 0..1023 ; 8 x 16B chunks per 64-f16 row
    int r = c >> 3;
    int q = c & 7;
#if USE_ASYNC
    __builtin_amdgcn_global_load_async_to_lds_b128(GPTR(gA + r * ldg + q),
                                                   LPTR(sA + r * LDSS + (q << 3)), 0, 0);
    __builtin_amdgcn_global_load_async_to_lds_b128(GPTR(gB + r * ldg + q),
                                                   LPTR(sB + r * LDSS + (q << 3)), 0, 0);
#else
    *reinterpret_cast<uint4*>(sA + r * LDSS + (q << 3)) = gA[r * ldg + q];
    *reinterpret_cast<uint4*>(sB + r * LDSS + (q << 3)) = gB[r * ldg + q];
#endif
  }
}

// ---------------- WMMA GEMM: C[M,N] = A[M,K] @ B[N,K]^T ----------------
template<bool GELU_F16>
__global__ __launch_bounds__(256)
void k_gemm(const _Float16* __restrict__ A, const _Float16* __restrict__ B,
            void* __restrict__ C, int M, int N, int K) {
  __shared__ _Float16 smem[2][2][BM * LDSS];  // [stage][A/B][tile]

  const int tid  = threadIdx.x;
  const int lane = tid & 31;
  const int wave = tid >> 5;
  const int wm   = wave >> 2;   // 0..1 : wave row (64 rows)
  const int wn   = wave & 3;    // 0..3 : wave col (32 cols)
  const int m0   = blockIdx.x * BM;
  const int n0   = blockIdx.y * BN;

  const v8f vzero = {0.f,0.f,0.f,0.f,0.f,0.f,0.f,0.f};
  v8f acc[4][2];
#pragma unroll
  for (int i = 0; i < 4; i++)
#pragma unroll
    for (int j = 0; j < 2; j++) acc[i][j] = vzero;

  // ISA 16-bit fragment addressing: lanes 0-15 -> K 0..7 & 16..23,
  // lanes 16-31 -> K 8..15 & 24..31 ; matrix row/col = lane & 15.
  const int frow = lane & 15;
  const int fko  = (lane >> 4) << 3;
  const int ntiles = K / BK;

  // prologue: stage tile 0
  copy_tiles(A, B, smem[0][0], smem[0][1], m0, n0, 0, K, tid);
  async_fence();
  __syncthreads();

  for (int t = 0; t < ntiles; t++) {
    const int cur = t & 1;
    if (t + 1 < ntiles)  // prefetch next tile into the other stage
      copy_tiles(A, B, smem[cur ^ 1][0], smem[cur ^ 1][1], m0, n0, (t + 1) * BK, K, tid);

    const _Float16* sA = smem[cur][0];
    const _Float16* sB = smem[cur][1];
#pragma unroll
    for (int s = 0; s < 2; s++) {   // two K=32 slices per stage
      v16h af[4], bfm[2];
#pragma unroll
      for (int i = 0; i < 4; i++) {
        const _Float16* p = sA + (wm * 64 + i * 16 + frow) * LDSS + s * 32 + fko;
        v8h lo = *reinterpret_cast<const v8h*>(p);
        v8h hi = *reinterpret_cast<const v8h*>(p + 16);
        af[i] = __builtin_shufflevector(lo, hi, 0,1,2,3,4,5,6,7,8,9,10,11,12,13,14,15);
      }
#pragma unroll
      for (int j = 0; j < 2; j++) {
        const _Float16* p = sB + (wn * 32 + j * 16 + frow) * LDSS + s * 32 + fko;
        v8h lo = *reinterpret_cast<const v8h*>(p);
        v8h hi = *reinterpret_cast<const v8h*>(p + 16);
        bfm[j] = __builtin_shufflevector(lo, hi, 0,1,2,3,4,5,6,7,8,9,10,11,12,13,14,15);
      }
#pragma unroll
      for (int i = 0; i < 4; i++)
#pragma unroll
        for (int j = 0; j < 2; j++)
          acc[i][j] = __builtin_amdgcn_wmma_f32_16x16x32_f16(
              false, af[i], false, bfm[j], (short)0, acc[i][j], false, false);
    }

    async_fence();     // next tile fully landed in LDS
    __syncthreads();   // all waves done reading cur / writing next
  }

  // Epilogue. C/D layout: lane<16 -> rows m..m+7, lane>=16 -> rows m+8..m+15; col = lane&15.
  const int cnb = n0 + wn * 32 + (lane & 15);
  const int rmb = m0 + wm * 64 + ((lane >> 4) << 3);
#pragma unroll
  for (int i = 0; i < 4; i++) {
#pragma unroll
    for (int j = 0; j < 2; j++) {
      const int cn = cnb + j * 16;
      const int rm = rmb + i * 16;
#pragma unroll
      for (int r = 0; r < 8; r++) {
        float v = acc[i][j][r];
        size_t idx = (size_t)(rm + r) * N + cn;
        if (GELU_F16) {
          reinterpret_cast<_Float16*>(C)[idx] = (_Float16)fast_gelu(v);
        } else {
          reinterpret_cast<float*>(C)[idx] = v;
        }
      }
    }
  }
}

extern "C" void kernel_launch(void* const* d_in, const int* in_sizes, int n_in,
                              void* d_out, int out_size, void* d_ws, size_t ws_size,
                              hipStream_t stream) {
  const float* x          = (const float*)d_in[0];
  const float* up_wq      = (const float*)d_in[1];
  const float* up_scale   = (const float*)d_in[2];
  const float* down_wq    = (const float*)d_in[3];
  const float* down_scale = (const float*)d_in[4];
  const float* up_signs   = (const float*)d_in[5];
  const float* down_signs = (const float*)d_in[6];

  char* ws = (char*)d_ws;
  _Float16* xh  = (_Float16*)ws;                                            // 32 MB
  _Float16* wup = (_Float16*)(ws + (size_t)T_TOK * D_IN * 2);               //  8 MB
  _Float16* wdn = (_Float16*)(ws + (size_t)T_TOK * D_IN * 2
                                 + (size_t)D_MID * D_IN * 2);               //  8 MB
  _Float16* act = (_Float16*)(ws + (size_t)T_TOK * D_IN * 2
                                 + (size_t)D_MID * D_IN * 2
                                 + (size_t)D_OUT * D_MID * 2);              // 128 MB

  // 1) x -> f16
  k_cvt_f16<<<(T_TOK * D_IN) / (4 * 256), 256, 0, stream>>>(x, xh);

  // 2) fold RHT + signs + scale into f16 weights (FWHT butterflies)
  k_prep_w<<<D_MID * (D_IN  / 128), 128, 0, stream>>>(up_wq,   up_scale,   up_signs,   wup, D_IN);
  k_prep_w<<<D_OUT * (D_MID / 128), 128, 0, stream>>>(down_wq, down_scale, down_signs, wdn, D_MID);

  // 3) h = x @ Wup^T, fused GELU, f16 activations
  dim3 g1(T_TOK / BM, D_MID / BN);
  k_gemm<true><<<g1, 256, 0, stream>>>(xh, wup, act, T_TOK, D_MID, D_IN);

  // 4) out = a @ Wdn^T, f32 output
  dim3 g2(T_TOK / BM, D_OUT / BN);
  k_gemm<false><<<g2, 256, 0, stream>>>(act, wdn, d_out, T_TOK, D_OUT, D_MID);
}
